// CausalSelfAttention_35639638622863
// MI455X (gfx1250) — matrix operand order repaired and checked by reading, not compiled, not run on hardware
//
#include <hip/hip_runtime.h>

// ---------------------------------------------------------------------------
// CausalSelfAttention for MI455X (gfx1250, wave32, WMMA bf16)
//   B=8, T=1024, C=1024, H=16, D=64
// Pipeline (all bf16 data / f32 accum via v_wmma_f32_16x16x32_bf16):
//   1) convert x -> bf16 ; LDS-tiled transpose W_qkv, W_out -> bf16 [N][K]
//   2) QKV GEMM (32x64 tiles, k double-buffered) -> Q,K [B,H,T,D], Vt [B,H,D,T]
//   3) flash attention (S^T = K Q^T trick; P tile re-packs as A operand)
//   4) output GEMM (32x64 tiles, k double-buffered) -> d_out (f32) + bias
// GEMM loops issue ALL loads of two K-steps before 16 back-to-back WMMAs so
// the allocator keeps fragment sets in distinct registers (no per-WMMA
// s_wait_loadcnt 0 drains).
// ---------------------------------------------------------------------------

typedef __attribute__((ext_vector_type(16))) __bf16 v16bf;
typedef __attribute__((ext_vector_type(8)))  float  v8f;
typedef __attribute__((ext_vector_type(4)))  unsigned int v4u;
typedef __attribute__((ext_vector_type(4)))  float  v4f;

union Frag {
    v16bf v;
    v4u   u4[2];
    unsigned short us[16];
};

__device__ __forceinline__ unsigned short f2bf(float f) {
    unsigned int u = __builtin_bit_cast(unsigned int, f);
    unsigned int r = u + 0x7FFFu + ((u >> 16) & 1u);   // round-to-nearest-even
    return (unsigned short)(r >> 16);
}

__device__ __forceinline__ v8f wmma_bf16(const Frag& a, const Frag& b, v8f c) {
    return __builtin_amdgcn_wmma_f32_16x16x32_bf16(false, a.v, false, b.v,
                                                   (short)0, c, false, false);
}

// A fragment: 16-bit A-layout = two 8-element runs at +8h and +16+8h
__device__ __forceinline__ void load_afrag(const unsigned short* row, int k, int half, Frag& a) {
    a.u4[0] = *(const v4u*)(row + k + 8 * half);
    a.u4[1] = *(const v4u*)(row + k + 16 + 8 * half);
}
// B fragment: two 8-element runs at the lane's [n][k + 8h] and [n][k + 16 + 8h]
__device__ __forceinline__ void load_bfrag(const unsigned short* p, Frag& b) {
    b.u4[0] = *(const v4u*)(p);
    b.u4[1] = *(const v4u*)(p + 16);
}

#define B_  8
#define T_  1024
#define C_  1024
#define H_  16
#define D_  64
#define N3_ 3072

// ---------------------------------------------------------------- converts --
__global__ void __launch_bounds__(256) cvt_x_kernel(const float* __restrict__ x,
                                                    unsigned short* __restrict__ x16) {
    const int i = (blockIdx.x * 256 + threadIdx.x) * 4;   // 8M elements, 4/thread
    const v4f f = *(const v4f*)(x + i);
    unsigned long long p = (unsigned long long)f2bf(f.x) |
                           ((unsigned long long)f2bf(f.y) << 16) |
                           ((unsigned long long)f2bf(f.z) << 32) |
                           ((unsigned long long)f2bf(f.w) << 48);
    *(unsigned long long*)(x16 + i) = p;
}

// W[k][n] (f32, row-major K x N) -> Wt[n][k] (bf16), 32x32 LDS tiles
__global__ void __launch_bounds__(256) transpose_w_kernel(const float* __restrict__ w,
                                                          unsigned short* __restrict__ wt,
                                                          int K, int N) {
    __shared__ unsigned short tile[32][33];               // +1 pad: no bank conflicts
    const int kb = blockIdx.x * 32, nb = blockIdx.y * 32;
    const int tx = threadIdx.x & 31, ty = threadIdx.x >> 5;   // ty: 0..7
#pragma unroll
    for (int i = ty; i < 32; i += 8)
        tile[i][tx] = f2bf(w[(size_t)(kb + i) * N + nb + tx]);
    __syncthreads();
#pragma unroll
    for (int i = ty; i < 32; i += 8)
        wt[(size_t)(nb + i) * K + kb + tx] = tile[tx][i];
}

// ---------------------------------------------------------------- QKV GEMM --
// one wave = one 32x64 output tile of [8192 x 3072]; two K=32 steps in flight
__global__ void __launch_bounds__(128) qkv_gemm_kernel(const unsigned short* __restrict__ X,
                                                       const unsigned short* __restrict__ Wt,
                                                       const float* __restrict__ bias,
                                                       unsigned short* __restrict__ Qb,
                                                       unsigned short* __restrict__ Kb,
                                                       unsigned short* __restrict__ Vt) {
    const int wave = blockIdx.x * 4 + (threadIdx.x >> 5);
    const int lane = threadIdx.x & 31;
    const int q    = lane & 15;
    const int half = lane >> 4;
    const int rt   = wave / (N3_ / 64);               // 256 row tiles (32 rows)
    const int ct   = wave % (N3_ / 64);               // 48 col tiles (64 cols)

    const unsigned short* ar0 = X + (size_t)(rt * 32 + q) * C_;
    const unsigned short* ar1 = ar0 + (size_t)16 * C_;
    const unsigned short* wb[4];
#pragma unroll
    for (int j = 0; j < 4; ++j)
        wb[j] = Wt + (size_t)(ct * 64 + j * 16 + q) * C_ + 8 * half;

    v8f acc[2][4];
#pragma unroll
    for (int m = 0; m < 2; ++m)
#pragma unroll
        for (int j = 0; j < 4; ++j)
            acc[m][j] = (v8f){0.f,0.f,0.f,0.f,0.f,0.f,0.f,0.f};

    for (int k0 = 0; k0 < C_; k0 += 64) {
        __builtin_prefetch(ar0 + k0 + 256, 0, 2);
        __builtin_prefetch(ar1 + k0 + 256, 0, 2);
        Frag a0[2], a1[2], b0[4], b1[4];
        load_afrag(ar0, k0, half, a0[0]);
        load_afrag(ar1, k0, half, a0[1]);
#pragma unroll
        for (int j = 0; j < 4; ++j) load_bfrag(wb[j] + k0, b0[j]);
        load_afrag(ar0, k0 + 32, half, a1[0]);
        load_afrag(ar1, k0 + 32, half, a1[1]);
#pragma unroll
        for (int j = 0; j < 4; ++j) load_bfrag(wb[j] + k0 + 32, b1[j]);

#pragma unroll
        for (int j = 0; j < 4; ++j) acc[0][j] = wmma_bf16(a0[0], b0[j], acc[0][j]);
#pragma unroll
        for (int j = 0; j < 4; ++j) acc[1][j] = wmma_bf16(a0[1], b0[j], acc[1][j]);
#pragma unroll
        for (int j = 0; j < 4; ++j) acc[0][j] = wmma_bf16(a1[0], b1[j], acc[0][j]);
#pragma unroll
        for (int j = 0; j < 4; ++j) acc[1][j] = wmma_bf16(a1[1], b1[j], acc[1][j]);
    }

    // scatter into Q [B,H,T,D] / K [B,H,T,D] / V^T [B,H,D,T]
#pragma unroll
    for (int j = 0; j < 4; ++j) {
        const int n  = ct * 64 + j * 16 + q;
        const float bv = bias[n];
        const int sel = n >> 10;                      // 0:Q 1:K 2:V
        const int nn  = n & (C_ - 1);
        const int h   = nn >> 6, d = nn & 63;
#pragma unroll
        for (int m = 0; m < 2; ++m)
#pragma unroll
            for (int r = 0; r < 8; ++r) {
                const int grow = rt * 32 + m * 16 + r + 8 * half;
                const int bb = grow >> 10, t = grow & (T_ - 1);
                const unsigned short val = f2bf(acc[m][j][r] + bv);
                if (sel == 0)
                    Qb[(((size_t)(bb * H_ + h) * T_) + t) * D_ + d] = val;
                else if (sel == 1)
                    Kb[(((size_t)(bb * H_ + h) * T_) + t) * D_ + d] = val;
                else
                    Vt[(((size_t)(bb * H_ + h) * D_) + d) * T_ + t] = val;
            }
    }
}

// ----------------------------------------------------------- flash attention
// one wave = one (b,h, 16-query tile). S^T = K*Q^T so that the softmaxed
// probability tile's C-layout IS the A-layout of the P*V WMMA (no LDS needed).
__global__ void __launch_bounds__(128) attn_kernel(const unsigned short* __restrict__ Qb,
                                                   const unsigned short* __restrict__ Kb,
                                                   const unsigned short* __restrict__ Vt,
                                                   unsigned short* __restrict__ AT) {
    const int wave = blockIdx.x * 4 + (threadIdx.x >> 5);
    const int lane = threadIdx.x & 31;
    const int qcol = lane & 15;
    const int half = lane >> 4;
    const int bh   = wave >> 6;                       // 0..127  (b*H+h)
    const int q0   = (wave & 63) * 16;

    const unsigned short* Qp = Qb + (size_t)bh * T_ * D_;
    const unsigned short* Kp = Kb + (size_t)bh * T_ * D_;
    const unsigned short* Vp = Vt + (size_t)bh * D_ * T_;

    // Q tile as B operands of S^T = K Q^T   (B[d,q] = Q[q][d])
    Frag bq[2];
#pragma unroll
    for (int c = 0; c < 2; ++c)
        load_bfrag(Qp + (size_t)(q0 + qcol) * D_ + 32 * c + 8 * half, bq[c]);

    v8f o[4];
#pragma unroll
    for (int j = 0; j < 4; ++j) o[j] = (v8f){0.f,0.f,0.f,0.f,0.f,0.f,0.f,0.f};
    float m = -1e30f, l = 0.f;

    const int ktmax = (q0 + 15) >> 5;                 // causal bound (32-key tiles)
    for (int kt = 0; kt <= ktmax; ++kt) {
        const int k0 = kt * 32;

        // -- issue ALL loads for this key tile up front: K A-fragments and
        //    V B-fragments. V latency hides under QK^T WMMAs + softmax VALU.
        Frag ak[2][2];                                 // [key 16-tile][D slab]
#pragma unroll
        for (int i = 0; i < 2; ++i)
#pragma unroll
            for (int c = 0; c < 2; ++c) {
                const unsigned short* p =
                    Kp + (size_t)(k0 + i * 16 + qcol) * D_ + 32 * c + 8 * half;
                ak[i][c].u4[0] = *(const v4u*)p;
                ak[i][c].u4[1] = *(const v4u*)(p + 16);
            }
        Frag bv[4];                                    // V: B[k,d] = Vt[d][k]
#pragma unroll
        for (int j = 0; j < 4; ++j)
            load_bfrag(Vp + (size_t)(j * 16 + qcol) * T_ + k0 + 8 * half, bv[j]);

        // S^T tiles: rows = keys, cols = queries
        v8f st[2];
#pragma unroll
        for (int i = 0; i < 2; ++i) {
            v8f s = (v8f){0.f,0.f,0.f,0.f,0.f,0.f,0.f,0.f};
            s = wmma_bf16(ak[i][0], bq[0], s);
            s = wmma_bf16(ak[i][1], bq[1], s);
            st[i] = s;
        }

        // scale + causal mask + local max over this wave's key subset
        float mloc = -1e30f;
#pragma unroll
        for (int i = 0; i < 2; ++i)
#pragma unroll
            for (int r = 0; r < 8; ++r) {
                const int key = k0 + i * 16 + r + 8 * half;
                float sv = st[i][r] * 0.125f;         // 1/sqrt(64)
                if (key > q0 + qcol) sv = -1e30f;
                st[i][r] = sv;
                mloc = fmaxf(mloc, sv);
            }
        mloc = fmaxf(mloc, __shfl_xor(mloc, 16, 32)); // combine key halves
        const float mnew  = fmaxf(m, mloc);
        const float alpha = __expf(m - mnew);

        // exponentiate; C-layout of S^T re-packs directly into the A fragment
        Frag pa;
        float ps = 0.f;
#pragma unroll
        for (int i = 0; i < 2; ++i)
#pragma unroll
            for (int r = 0; r < 8; ++r) {
                const float p = __expf(st[i][r] - mnew);
                ps += p;
                pa.us[i * 8 + r] = f2bf(p);
            }
        ps += __shfl_xor(ps, 16, 32);
        l = l * alpha + ps;
        m = mnew;

        // rescale accumulators (row r+8*half of O == query r+8*half)
#pragma unroll
        for (int r = 0; r < 8; ++r) {
            const float a8 = __shfl(alpha, half * 8 + r, 32);
#pragma unroll
            for (int j = 0; j < 4; ++j) o[j][r] *= a8;
        }

        // O += P * V  (V fragments already in registers)
#pragma unroll
        for (int j = 0; j < 4; ++j)
            o[j] = wmma_bf16(pa, bv[j], o[j]);
    }

    // normalize and store to [B,T,C] bf16
    const int b = bh >> 4, h = bh & 15;
#pragma unroll
    for (int r = 0; r < 8; ++r) {
        const float linv = 1.f / __shfl(l, half * 8 + r, 32);
        const int t = q0 + r + 8 * half;
        unsigned short* outp = AT + ((size_t)(b * T_ + t) * C_) + h * D_;
#pragma unroll
        for (int j = 0; j < 4; ++j)
            outp[j * 16 + qcol] = f2bf(o[j][r] * linv);
    }
}

// ------------------------------------------------------------- output GEMM --
// one wave = one 32x64 output tile of [8192 x 1024]; two K=32 steps in flight
__global__ void __launch_bounds__(128) out_gemm_kernel(const unsigned short* __restrict__ A,
                                                       const unsigned short* __restrict__ Wt,
                                                       const float* __restrict__ bias,
                                                       float* __restrict__ out) {
    const int wave = blockIdx.x * 4 + (threadIdx.x >> 5);
    const int lane = threadIdx.x & 31;
    const int q    = lane & 15;
    const int half = lane >> 4;
    const int rt   = wave / (C_ / 64);                // 256 row tiles (32 rows)
    const int ct   = wave % (C_ / 64);                // 16 col tiles

    const unsigned short* ar0 = A + (size_t)(rt * 32 + q) * C_;
    const unsigned short* ar1 = ar0 + (size_t)16 * C_;
    const unsigned short* wb[4];
#pragma unroll
    for (int j = 0; j < 4; ++j)
        wb[j] = Wt + (size_t)(ct * 64 + j * 16 + q) * C_ + 8 * half;

    v8f acc[2][4];
#pragma unroll
    for (int m = 0; m < 2; ++m)
#pragma unroll
        for (int j = 0; j < 4; ++j)
            acc[m][j] = (v8f){0.f,0.f,0.f,0.f,0.f,0.f,0.f,0.f};

    for (int k0 = 0; k0 < C_; k0 += 64) {
        __builtin_prefetch(ar0 + k0 + 256, 0, 2);
        __builtin_prefetch(ar1 + k0 + 256, 0, 2);
        Frag a0[2], a1[2], b0[4], b1[4];
        load_afrag(ar0, k0, half, a0[0]);
        load_afrag(ar1, k0, half, a0[1]);
#pragma unroll
        for (int j = 0; j < 4; ++j) load_bfrag(wb[j] + k0, b0[j]);
        load_afrag(ar0, k0 + 32, half, a1[0]);
        load_afrag(ar1, k0 + 32, half, a1[1]);
#pragma unroll
        for (int j = 0; j < 4; ++j) load_bfrag(wb[j] + k0 + 32, b1[j]);

#pragma unroll
        for (int j = 0; j < 4; ++j) acc[0][j] = wmma_bf16(a0[0], b0[j], acc[0][j]);
#pragma unroll
        for (int j = 0; j < 4; ++j) acc[1][j] = wmma_bf16(a0[1], b0[j], acc[1][j]);
#pragma unroll
        for (int j = 0; j < 4; ++j) acc[0][j] = wmma_bf16(a1[0], b1[j], acc[0][j]);
#pragma unroll
        for (int j = 0; j < 4; ++j) acc[1][j] = wmma_bf16(a1[1], b1[j], acc[1][j]);
    }

#pragma unroll
    for (int j = 0; j < 4; ++j) {
        const int n = ct * 64 + j * 16 + q;
        const float bv = bias[n];
#pragma unroll
        for (int m = 0; m < 2; ++m)
#pragma unroll
            for (int r = 0; r < 8; ++r) {
                const int grow = rt * 32 + m * 16 + r + 8 * half;
                out[(size_t)grow * C_ + n] = acc[m][j][r] + bv;
            }
    }
}

// ------------------------------------------------------------------ launch --
extern "C" void kernel_launch(void* const* d_in, const int* in_sizes, int n_in,
                              void* d_out, int out_size, void* d_ws, size_t ws_size,
                              hipStream_t stream) {
    const float* x     = (const float*)d_in[0];
    const float* W_qkv = (const float*)d_in[1];
    const float* b_qkv = (const float*)d_in[2];
    const float* W_out = (const float*)d_in[3];
    const float* b_out = (const float*)d_in[4];
    float* out = (float*)d_out;

    // workspace layout (bytes)
    char* ws = (char*)d_ws;
    const size_t SZ_X16  = (size_t)B_ * T_ * C_ * 2;     // 16 MB
    const size_t SZ_WQT  = (size_t)N3_ * C_ * 2;         //  6 MB
    const size_t SZ_WOT  = (size_t)C_ * C_ * 2;          //  2 MB
    const size_t SZ_QKV  = (size_t)B_ * T_ * C_ * 2;     // 16 MB each
    unsigned short* X16  = (unsigned short*)(ws);
    unsigned short* WQT  = (unsigned short*)(ws + SZ_X16);
    unsigned short* WOT  = (unsigned short*)(ws + SZ_X16 + SZ_WQT);
    unsigned short* Qb   = (unsigned short*)(ws + SZ_X16 + SZ_WQT + SZ_WOT);
    unsigned short* Kb   = (unsigned short*)((char*)Qb + SZ_QKV);
    unsigned short* Vt   = (unsigned short*)((char*)Kb + SZ_QKV);
    unsigned short* AT   = (unsigned short*)((char*)Vt + SZ_QKV);

    // 1) precision conversion / weight transposes
    cvt_x_kernel<<<(B_ * T_ * C_) / (256 * 4), 256, 0, stream>>>(x, X16);
    {
        dim3 g1(C_ / 32, N3_ / 32);
        transpose_w_kernel<<<g1, 256, 0, stream>>>(W_qkv, WQT, C_, N3_);
        dim3 g2(C_ / 32, C_ / 32);
        transpose_w_kernel<<<g2, 256, 0, stream>>>(W_out, WOT, C_, C_);
    }

    // 2) QKV projection: 256x48 wave tiles (32x64 each), 4 waves/block
    qkv_gemm_kernel<<<(256 * 48) / 4, 128, 0, stream>>>(X16, WQT, b_qkv, Qb, Kb, Vt);

    // 3) flash attention: 8 * 16 * 64 = 8192 wave tiles
    attn_kernel<<<8192 / 4, 128, 0, stream>>>(Qb, Kb, Vt, AT);

    // 4) output projection: 256x16 wave tiles (32x64 each)
    out_gemm_kernel<<<(256 * 16) / 4, 128, 0, stream>>>(AT, WOT, b_out, out);
}
